// SlotAttention_5007931867556
// MI455X (gfx1250) — matrix operand order repaired and checked by reading, not compiled, not run on hardware
//
#include <hip/hip_runtime.h>
#include <hip/hip_bf16.h>
#include <math.h>

// ---------------- problem constants ----------------
#define BB     64
#define NN     4096
#define DD     256
#define NSLOT  8
#define HH     256
#define ITERS_ 10
#define EPS_   1e-8f
#define LN_EPS_ 1e-5f
#define SCALE_ 0.0625f   // 256^-0.5
#define KSPLIT 8         // split-K factor for updates GEMM (4096/8 = 512 per chunk)

// ---------------- WMMA types / helpers ----------------
typedef __attribute__((ext_vector_type(16))) __bf16        v16bf;
typedef __attribute__((ext_vector_type(8)))  float         v8f;
typedef __attribute__((ext_vector_type(4)))  unsigned int  u32x4;
typedef __attribute__((ext_vector_type(8)))  unsigned int  u32x8;

union BFragU { u32x8 u; v16bf v; };

__device__ __forceinline__ unsigned short f2bf(float f) {
  unsigned int u = __float_as_uint(f);
  u += 0x7FFFu + ((u >> 16) & 1u);           // round-to-nearest-even
  return (unsigned short)(u >> 16);
}

// A-matrix 16x32 bf16 fragment: lanes 0-15 hold K=k0+[0..15],
// lanes 16-31 hold K=k0+[16..31]; row M = lane%16. rowp = ptr to this lane's row.
__device__ __forceinline__ v16bf ldA(const unsigned short* rowp, int k0, int lane) {
  int off = (lane < 16) ? 0 : 16;
  u32x4 c0 = *(const u32x4*)(rowp + k0 + off);
  u32x4 c1 = *(const u32x4*)(rowp + k0 + off + 8);
  BFragU f; f.u = (u32x8){c0.x, c0.y, c0.z, c0.w, c1.x, c1.y, c1.z, c1.w};
  return f.v;
}
// B-matrix 32x16 bf16 fragment: lane j holds column N=j%16; lanes 0-15 K=k0..k0+15,
// lanes 16-31 K=k0+16..k0+31. colp = ptr to this lane's column (contiguous over K).
__device__ __forceinline__ v16bf ldB(const unsigned short* colp, int k0, int lane) {
  int off = (lane < 16) ? 0 : 16;
  BFragU f; f.u = *(const u32x8*)(colp + k0 + off);
  return f.v;
}

#define WMMA_BF16(a, b, c) \
  __builtin_amdgcn_wmma_f32_16x16x32_bf16(false, (a), false, (b), (short)0, (c), false, false)

// ---------------- one-time init: slots = mu + sigma*noise; zero pad rows; bf16 weights ----
__global__ void init_kernel(const float* __restrict__ noise, const float* __restrict__ mu,
                            const float* __restrict__ sigma,
                            const float* __restrict__ Wk, const float* __restrict__ Wv,
                            float* __restrict__ slots, unsigned short* __restrict__ qbf,
                            unsigned short* __restrict__ attn,
                            unsigned short* __restrict__ wkbf, unsigned short* __restrict__ wvbf) {
  long idx = (long)blockIdx.x * 256 + threadIdx.x;           // grid covers BB*8*NN
  if (idx < (long)BB * NSLOT * DD) {
    int d = (int)(idx & (DD - 1));
    slots[idx] = mu[d] + sigma[d] * noise[idx];
    long b = idx / (NSLOT * DD), r = idx % (NSLOT * DD);     // zero q rows 8..15
    qbf[b * 16 * DD + NSLOT * DD + r] = 0;
  }
  if (idx < (long)DD * DD) { wkbf[idx] = f2bf(Wk[idx]); wvbf[idx] = f2bf(Wv[idx]); }
  if (idx < (long)BB * NSLOT * NN) {                         // zero attn rows 8..15
    long b = idx / (NSLOT * NN), r = idx % (NSLOT * NN);
    attn[b * 16 * NN + NSLOT * NN + r] = 0;
  }
}

// ---------------- LN(inputs) -> bf16, one wave32 per row ----------------
__global__ void ln_in_kernel(const float* __restrict__ x, const float* __restrict__ g,
                             const float* __restrict__ bb, unsigned short* __restrict__ y) {
  int wid = threadIdx.x >> 5, lane = threadIdx.x & 31;
  long row = (long)blockIdx.x * 8 + wid;
  const float* xr = x + row * DD + lane * 8;
  float v[8]; float s = 0.f, q2 = 0.f;
#pragma unroll
  for (int j = 0; j < 8; j++) { v[j] = xr[j]; s += v[j]; q2 += v[j] * v[j]; }
#pragma unroll
  for (int m = 16; m >= 1; m >>= 1) { s += __shfl_xor(s, m, 32); q2 += __shfl_xor(q2, m, 32); }
  float mean = s * (1.f / DD);
  float rstd = rsqrtf(q2 * (1.f / DD) - mean * mean + LN_EPS_);
  unsigned short o[8];
#pragma unroll
  for (int j = 0; j < 8; j++)
    o[j] = f2bf((v[j] - mean) * rstd * g[lane * 8 + j] + bb[lane * 8 + j]);
  u32x4 w = { (unsigned)o[0] | ((unsigned)o[1] << 16), (unsigned)o[2] | ((unsigned)o[3] << 16),
              (unsigned)o[4] | ((unsigned)o[5] << 16), (unsigned)o[6] | ((unsigned)o[7] << 16) };
  *(u32x4*)(y + row * DD + lane * 8) = w;
}

// ---------------- K = xln@Wk^T+bk ; V^T stored [B,D,N] ; WMMA bf16 ----------------
__global__ void kv_kernel(const unsigned short* __restrict__ xln,
                          const unsigned short* __restrict__ wk, const float* __restrict__ bk,
                          const unsigned short* __restrict__ wv, const float* __restrict__ bv,
                          unsigned short* __restrict__ kout, unsigned short* __restrict__ vT) {
  int wid = threadIdx.x >> 5, lane = threadIdx.x & 31;
  int gid = blockIdx.x * 8 + wid;
  int mtile = gid >> 4, dtile = gid & 15;
  long m_base = (long)mtile * 16;
  int d_base = dtile * 16, l15 = lane & 15;
  const unsigned short* ap  = xln + (m_base + l15) * DD;
  const unsigned short* bkp = wk + (long)(d_base + l15) * DD;   // Wk row d_out = B column
  const unsigned short* bvp = wv + (long)(d_base + l15) * DD;
  v8f ck = {}; v8f cv = {};
#pragma unroll
  for (int kk = 0; kk < DD; kk += 32) {
    v16bf a = ldA(ap, kk, lane);
    ck = WMMA_BF16(a, ldB(bkp, kk, lane), ck);
    cv = WMMA_BF16(a, ldB(bvp, kk, lane), cv);
  }
  float bkv = bk[d_base + l15], bvv = bv[d_base + l15];
  int rbase = (lane < 16) ? 0 : 8;
#pragma unroll
  for (int g2 = 0; g2 < 8; g2++) {
    long row = m_base + rbase + g2;                  // global token index
    int b = (int)(row >> 12), n = (int)(row & (NN - 1));
    kout[row * DD + d_base + l15] = f2bf(ck[g2] + bkv);
    vT[((long)b * DD + d_base + l15) * NN + n] = f2bf(cv[g2] + bvv);
  }
}

// ---- per-iter: slot LN + q (scalar, tiny); also zero rowsum and updates accumulator ----
__global__ void qln_kernel(const float* __restrict__ slots, const float* __restrict__ g,
                           const float* __restrict__ bb, const float* __restrict__ Wq,
                           const float* __restrict__ bq, unsigned short* __restrict__ qbf,
                           float* __restrict__ rowsum, float* __restrict__ updates) {
  __shared__ float sn[NSLOT * DD];
  __shared__ float red1[8], red2[8];
  int t = threadIdx.x, b = blockIdx.x, lane = t & 31, wid = t >> 5;
  if (t < 16) rowsum[b * 16 + t] = 0.f;
  {
    float* up = updates + (long)b * NSLOT * DD;     // zero this batch's update accumulator
#pragma unroll
    for (int j = 0; j < NSLOT; j++) up[j * DD + t] = 0.f;
  }
  for (int i = 0; i < NSLOT; i++) {
    float v = slots[((long)b * NSLOT + i) * DD + t];
    float s = v, q2 = v * v;
#pragma unroll
    for (int m = 16; m >= 1; m >>= 1) { s += __shfl_xor(s, m, 32); q2 += __shfl_xor(q2, m, 32); }
    if (lane == 0) { red1[wid] = s; red2[wid] = q2; }
    __syncthreads();
    float ts = 0.f, tq = 0.f;
#pragma unroll
    for (int j = 0; j < 8; j++) { ts += red1[j]; tq += red2[j]; }
    float mean = ts * (1.f / DD);
    sn[i * DD + t] = (v - mean) * rsqrtf(tq * (1.f / DD) - mean * mean + LN_EPS_) * g[t] + bb[t];
    __syncthreads();
  }
  for (int i = 0; i < NSLOT; i++) {
    float acc = bq[t];
    const float* wr = Wq + (long)t * DD;
    for (int d = 0; d < DD; d++) acc = fmaf(wr[d], sn[i * DD + d], acc);
    qbf[((long)b * 16 + i) * DD + t] = f2bf(acc);
  }
}

// ---------------- dots + softmax-over-slots + row-sum accumulation ----------------
__global__ void attn_kernel(const unsigned short* __restrict__ qbf,
                            const unsigned short* __restrict__ kbf,
                            unsigned short* __restrict__ attn, float* __restrict__ rowsum) {
  int wid = threadIdx.x >> 5, lane = threadIdx.x & 31;
  int gid = blockIdx.x * 8 + wid;
  int b = gid >> 8, n_base = (gid & 255) * 16, l15 = lane & 15;
  const unsigned short* ap = qbf + ((long)b * 16 + l15) * DD;
  const unsigned short* bp = kbf + ((long)b * NN + n_base + l15) * DD;   // k row j = B column
  v8f c = {};
#pragma unroll
  for (int kk = 0; kk < DD; kk += 32)
    c = WMMA_BF16(ldA(ap, kk, lane), ldB(bp, kk, lane), c);
  // lane j (0-15) holds dots[slot 0..7, n_base+j] in c[0..7]: softmax over slots in-lane.
  float d[8]; float mx = -1e30f;
#pragma unroll
  for (int g2 = 0; g2 < 8; g2++) { d[g2] = c[g2] * SCALE_; mx = fmaxf(mx, d[g2]); }
  float s = 0.f;
#pragma unroll
  for (int g2 = 0; g2 < 8; g2++) { d[g2] = __expf(d[g2] - mx); s += d[g2]; }
  float inv = 1.f / s;
  float p[8];
#pragma unroll
  for (int g2 = 0; g2 < 8; g2++) p[g2] = d[g2] * inv + EPS_;
  if (lane < 16) {
#pragma unroll
    for (int g2 = 0; g2 < 8; g2++)
      attn[((long)b * 16 + g2) * NN + n_base + lane] = f2bf(p[g2]);
  }
  float t[8];
#pragma unroll
  for (int g2 = 0; g2 < 8; g2++) {
    t[g2] = (lane < 16) ? p[g2] : 0.f;
#pragma unroll
    for (int m = 8; m >= 1; m >>= 1) t[g2] += __shfl_xor(t[g2], m, 32);   // sum lanes 0..15
  }
  if (lane == 0) {
#pragma unroll
    for (int g2 = 0; g2 < 8; g2++) atomicAdd(&rowsum[b * 16 + g2], t[g2]);
  }
}

// ---- updates += (attn/rowsum) @ v, split-K over KSPLIT chunks for occupancy ----
// 8192 waves instead of 1024: each handles a 512-wide K slice (16 WMMAs) and
// atomically accumulates its pre-scaled partial tile into the fp32 accumulator.
__global__ void upd_kernel(const unsigned short* __restrict__ attn,
                           const unsigned short* __restrict__ vT,
                           const float* __restrict__ rowsum, float* __restrict__ updates) {
  int wid = threadIdx.x >> 5, lane = threadIdx.x & 31;
  int gid = blockIdx.x * 8 + wid;                  // total BB*16*KSPLIT
  int ks   = gid & (KSPLIT - 1);
  int rest = gid / KSPLIT;
  int b = rest >> 4, d_base = (rest & 15) * 16, l15 = lane & 15;
  int k0 = ks * (NN / KSPLIT);
  const unsigned short* ap = attn + ((long)b * 16 + l15) * NN;
  const unsigned short* bp = vT + ((long)b * DD + d_base + l15) * NN;
  v8f c = {};
#pragma unroll 2
  for (int kk = k0; kk < k0 + NN / KSPLIT; kk += 32) {
    __builtin_prefetch(bp + kk + 512, 0, 1);
    c = WMMA_BF16(ldA(ap, kk, lane), ldB(bp, kk, lane), c);
  }
  int rbase = (lane < 16) ? 0 : 8;
#pragma unroll
  for (int g2 = 0; g2 < 8; g2++) c[g2] *= 1.f / rowsum[b * 16 + rbase + g2];
  if (lane < 16) {
#pragma unroll
    for (int g2 = 0; g2 < 8; g2++)
      atomicAdd(&updates[((long)b * NSLOT + g2) * DD + d_base + lane], c[g2]);
  }
}

// ---------------- GRUCell + residual MLP, one block per (b, slot) row ----------------
__global__ void gru_kernel(float* __restrict__ slots, const float* __restrict__ updates,
                           const float* __restrict__ Wih, const float* __restrict__ bih,
                           const float* __restrict__ Whh, const float* __restrict__ bhh,
                           const float* __restrict__ W1, const float* __restrict__ b1,
                           const float* __restrict__ W2, const float* __restrict__ b2,
                           const float* __restrict__ g, const float* __restrict__ bb) {
  __shared__ float su[DD], ss[DD], hbuf[DD], abuf[DD];
  __shared__ float red1[8], red2[8];
  int t = threadIdx.x; long row = blockIdx.x;
  su[t] = updates[row * DD + t]; ss[t] = slots[row * DD + t];
  __syncthreads();
  float xr = bih[t], xz = bih[DD + t], xn = bih[2 * DD + t];
  float hr = bhh[t], hz = bhh[DD + t], hn = bhh[2 * DD + t];
  const float* wr = Wih + (long)t * DD;            const float* vr = Whh + (long)t * DD;
  const float* wz = Wih + (long)(DD + t) * DD;     const float* vz = Whh + (long)(DD + t) * DD;
  const float* wn = Wih + (long)(2 * DD + t) * DD; const float* vn = Whh + (long)(2 * DD + t) * DD;
  for (int d = 0; d < DD; d++) {
    float u = su[d], s0 = ss[d];
    xr = fmaf(wr[d], u, xr); xz = fmaf(wz[d], u, xz); xn = fmaf(wn[d], u, xn);
    hr = fmaf(vr[d], s0, hr); hz = fmaf(vz[d], s0, hz); hn = fmaf(vn[d], s0, hn);
  }
  float r = 1.f / (1.f + __expf(-(xr + hr)));
  float z = 1.f / (1.f + __expf(-(xz + hz)));
  float n = tanhf(xn + r * hn);
  float h = (1.f - z) * n + z * ss[t];
  int lane = t & 31, wid = t >> 5;
  float s = h, q2 = h * h;
#pragma unroll
  for (int m = 16; m >= 1; m >>= 1) { s += __shfl_xor(s, m, 32); q2 += __shfl_xor(q2, m, 32); }
  if (lane == 0) { red1[wid] = s; red2[wid] = q2; }
  __syncthreads();
  float ts = 0.f, tq = 0.f;
#pragma unroll
  for (int j = 0; j < 8; j++) { ts += red1[j]; tq += red2[j]; }
  float mean = ts * (1.f / DD);
  hbuf[t] = (h - mean) * rsqrtf(tq * (1.f / DD) - mean * mean + LN_EPS_) * g[t] + bb[t];
  __syncthreads();
  float a = b1[t];
  const float* w1 = W1 + (long)t * DD;
  for (int d = 0; d < DD; d++) a = fmaf(w1[d], hbuf[d], a);
  abuf[t] = fmaxf(a, 0.f);
  __syncthreads();
  float o = b2[t];
  const float* w2 = W2 + (long)t * HH;
  for (int d = 0; d < HH; d++) o = fmaf(w2[d], abuf[d], o);
  slots[row * DD + t] = h + o;
}

__global__ void copy_kernel(const float* __restrict__ src, float* __restrict__ dst) {
  long i = (long)blockIdx.x * 256 + threadIdx.x;
  dst[i] = src[i];
}

// ---------------- host orchestration ----------------
extern "C" void kernel_launch(void* const* d_in, const int* in_sizes, int n_in,
                              void* d_out, int out_size, void* d_ws, size_t ws_size,
                              hipStream_t stream) {
  const float* inputs = (const float*)d_in[0];
  const float* noise  = (const float*)d_in[1];
  const float* mu     = (const float*)d_in[2];
  const float* sigma  = (const float*)d_in[3];
  const float* Wq  = (const float*)d_in[4];  const float* bq  = (const float*)d_in[5];
  const float* Wk  = (const float*)d_in[6];  const float* bk  = (const float*)d_in[7];
  const float* Wv  = (const float*)d_in[8];  const float* bv  = (const float*)d_in[9];
  const float* Wih = (const float*)d_in[10]; const float* bih = (const float*)d_in[11];
  const float* Whh = (const float*)d_in[12]; const float* bhh = (const float*)d_in[13];
  const float* W1  = (const float*)d_in[14]; const float* b1  = (const float*)d_in[15];
  const float* W2  = (const float*)d_in[16]; const float* b2  = (const float*)d_in[17];
  const float* ln_in_g = (const float*)d_in[18]; const float* ln_in_b = (const float*)d_in[19];
  const float* ln_s_g  = (const float*)d_in[20]; const float* ln_s_b  = (const float*)d_in[21];
  const float* ln_ff_g = (const float*)d_in[22]; const float* ln_ff_b = (const float*)d_in[23];

  char* p = (char*)d_ws;
  auto take = [&](size_t bytes) -> void* {
    void* r = (void*)p; p += (bytes + 255) & ~(size_t)255; return r;
  };
  float* slots   = (float*)take((size_t)BB * NSLOT * DD * 4);
  float* updates = (float*)take((size_t)BB * NSLOT * DD * 4);
  float* rowsum  = (float*)take((size_t)BB * 16 * 4);
  unsigned short* wkbf = (unsigned short*)take((size_t)DD * DD * 2);
  unsigned short* wvbf = (unsigned short*)take((size_t)DD * DD * 2);
  unsigned short* qbf  = (unsigned short*)take((size_t)BB * 16 * DD * 2);
  unsigned short* attn = (unsigned short*)take((size_t)BB * 16 * NN * 2);
  unsigned short* xln  = (unsigned short*)take((size_t)BB * NN * DD * 2);
  unsigned short* kbf  = (unsigned short*)take((size_t)BB * NN * DD * 2);
  unsigned short* vT   = (unsigned short*)take((size_t)BB * NN * DD * 2);

  init_kernel<<<(BB * NSLOT * NN) / 256, 256, 0, stream>>>(
      noise, mu, sigma, Wk, Wv, slots, qbf, attn, wkbf, wvbf);
  ln_in_kernel<<<(BB * NN) / 8, 256, 0, stream>>>(inputs, ln_in_g, ln_in_b, xln);
  kv_kernel<<<(BB * NN / 16) * 16 / 8, 256, 0, stream>>>(xln, wkbf, bk, wvbf, bv, kbf, vT);
  for (int it = 0; it < ITERS_; ++it) {
    qln_kernel<<<BB, 256, 0, stream>>>(slots, ln_s_g, ln_s_b, Wq, bq, qbf, rowsum, updates);
    attn_kernel<<<(BB * (NN / 16)) / 8, 256, 0, stream>>>(qbf, kbf, attn, rowsum);
    upd_kernel<<<(BB * 16 * KSPLIT) / 8, 256, 0, stream>>>(attn, vT, rowsum, updates);
    gru_kernel<<<BB * NSLOT, 256, 0, stream>>>(slots, updates, Wih, bih, Whh, bhh,
                                               W1, b1, W2, b2, ln_ff_g, ln_ff_b);
  }
  copy_kernel<<<(BB * NSLOT * DD) / 256, 256, 0, stream>>>(slots, (float*)d_out);
}
